// EdgeModel_31044023616095
// MI455X (gfx1250) — matrix operand order repaired and checked by reading, not compiled, not run on hardware
//
#include <hip/hip_runtime.h>
#include <hip/hip_bf16.h>

typedef __attribute__((ext_vector_type(16))) __bf16 v16bf;
typedef __attribute__((ext_vector_type(8)))  float  v8f;

#define ALPHA 0.8f

// ---------------------------------------------------------------------------
// 1) degree accumulation: deg[row[e]] += w[e]
__global__ __launch_bounds__(256) void deg_kernel(const int* __restrict__ row,
                                                  const float* __restrict__ w,
                                                  float* __restrict__ deg, int E) {
    int e = blockIdx.x * 256 + threadIdx.x;
    if (e < E) atomicAdd(&deg[row[e]], w[e]);
}

// 2) deg -> dinv in place, with GCN fixup deg<0.5 -> deg+1
__global__ __launch_bounds__(256) void dinv_kernel(float* __restrict__ deg, int n) {
    int i = blockIdx.x * 256 + threadIdx.x;
    if (i < n) {
        float d = deg[i];
        if (d < 0.5f) d += 1.0f;
        deg[i] = rsqrtf(d);
    }
}

// 3) SpMM: one wave (32 lanes) per edge, float4 per lane covers 128 features.
//    val = dinv[row]*w*dinv[col] computed on the fly; scatter with f32 atomics.
__global__ __launch_bounds__(256) void spmm_kernel(const int* __restrict__ rows,
                                                   const int* __restrict__ cols,
                                                   const float* __restrict__ w,
                                                   const float* __restrict__ dinv,
                                                   const float* __restrict__ X,
                                                   float* __restrict__ Mout, int E) {
    long long t = (long long)blockIdx.x * 256 + threadIdx.x;
    int e = (int)(t >> 5);
    int lane = threadIdx.x & 31;
    if (e >= E) return;
    int r = rows[e], c = cols[e];
    float val = dinv[r] * w[e] * dinv[c];
    float4 v = ((const float4*)(X + (long long)c * 128))[lane];
    float* mr = Mout + (long long)r * 128 + lane * 4;
    atomicAdd(mr + 0, val * v.x);
    atomicAdd(mr + 1, val * v.y);
    atomicAdd(mr + 2, val * v.z);
    atomicAdd(mr + 3, val * v.w);
}

// 4) Y[r,o] = sum_k X[r,k]*W[o,k] + bias[o]   (h = m @ W^T + b)
//    bf16 WMMA 16x16x32, f32 accumulate. 64-row X tile in LDS; W^T staged in LDS
//    in a fragment-contiguous swizzle so every B fragment is 2x ds_load_b128.
__global__ __launch_bounds__(256) void gemm_bias_wmma_kernel(const float* __restrict__ X,
                                                             const float* __restrict__ W,
                                                             const float* __restrict__ bias,
                                                             float* __restrict__ Y, int rows) {
    __shared__ __attribute__((aligned(16))) __bf16 sWf[128 * 128];  // swizzled W^T fragments
    __shared__ __attribute__((aligned(16))) __bf16 sX[64 * 128];    // 64-row input tile
    const int tid = threadIdx.x;

    // fill swizzled W fragments: 2048 groups of 8 contiguous bf16 (one b128 store each)
    for (int g = tid; g < 2048; g += 256) {
        int s    = g * 8;            // flat element index; j0 = s & 15 is 0 or 8
        int j0   = s & 15;
        int lane = (s >> 4) & 31;
        int tc   = (s >> 9) & 7;
        int kb   = s >> 12;
        int k    = kb * 32 + (lane >> 4) * 8 + j0 * 2;   // table-order k for j0..j0+7
        int o    = tc * 16 + (lane & 15);
        const float* wsrc = &W[o * 128 + k];
        __bf16* dst = &sWf[s];
#pragma unroll
        for (int j = 0; j < 8; ++j) dst[j] = (__bf16)wsrc[j];
    }

    const int r0 = blockIdx.x * 64;
    const bool full_tile = (r0 + 64 <= rows);   // block-uniform
    if (full_tile) {
        for (int idx = tid; idx < 64 * 128; idx += 256) {
            int r = r0 + (idx >> 7);
            sX[idx] = (__bf16)X[(long long)r * 128 + (idx & 127)];
        }
    } else {
        for (int idx = tid; idx < 64 * 128; idx += 256) {
            int r = r0 + (idx >> 7);
            sX[idx] = (__bf16)(r < rows ? X[(long long)r * 128 + (idx & 127)] : 0.0f);
        }
    }
    __syncthreads();

    const int wave = tid >> 5;
    const int lane = tid & 31;
    const int half = lane >> 4;         // 0: lanes 0-15, 1: lanes 16-31
    const int l15  = lane & 15;
    const int sr   = (wave >> 1) * 16;  // 16-row strip within tile

    // A fragments (16x32 bf16, ISA layout: lanes 0-15 hold K 0-7 & 16-23,
    // lanes 16-31 hold K 8-15 & 24-31, 2 values per VGPR) -> 2x b128 per frag
    v16bf afrag[4];
    {
        const __bf16* xr = &sX[(sr + l15) * 128];
#pragma unroll
        for (int kb = 0; kb < 4; ++kb) {
            const __bf16* p = xr + kb * 32;
#pragma unroll
            for (int i = 0; i < 4; ++i) {
                afrag[kb][2 * i]     = p[half * 8 + 2 * i];
                afrag[kb][2 * i + 1] = p[half * 8 + 2 * i + 1];
                afrag[kb][2 * i + 8] = p[16 + half * 8 + 2 * i];
                afrag[kb][2 * i + 9] = p[16 + half * 8 + 2 * i + 1];
            }
        }
    }

    const int gr0 = r0 + sr + 8 * half;  // D layout: VGPR v -> M = v + 8*half

#pragma unroll
    for (int ct = 0; ct < 4; ++ct) {
        const int tc = (wave & 1) * 4 + ct;   // 16-column tile index 0..7
        const int c0 = tc * 16;
        v8f acc = {};
#pragma unroll
        for (int kb = 0; kb < 4; ++kb) {
            const __bf16* wp = &sWf[((kb * 8 + tc) * 32 + lane) * 16];
            v16bf bfrag;
#pragma unroll
            for (int j = 0; j < 16; ++j) bfrag[j] = wp[j];  // 32 contiguous bytes -> 2x ds_load_b128
            acc = __builtin_amdgcn_wmma_f32_16x16x32_bf16(false, afrag[kb], false, bfrag,
                                                          (short)0, acc, false, false);
        }
        const float bv = bias[c0 + l15];
        float* yp = Y + (long long)gr0 * 128 + c0 + l15;  // one base addr per lane
        if (full_tile) {
            // scalar-uniform fast path: constant immediate offsets, no exec churn
#pragma unroll
            for (int v = 0; v < 8; ++v) yp[v * 128] = acc[v] + bv;
        } else {
#pragma unroll
            for (int v = 0; v < 8; ++v)
                if (gr0 + v < rows) yp[v * 128] = acc[v] + bv;
        }
    }
}

// 5) per-column sum / sumsq over the node dimension (512 rows per block)
__global__ __launch_bounds__(256) void colstats_kernel(const float* __restrict__ Y, int rows,
                                                       float* __restrict__ sum,
                                                       float* __restrict__ sumsq) {
    int tid = threadIdx.x;
    int col = tid & 127;
    int r0 = blockIdx.x * 512 + (tid >> 7);
    float s = 0.f, q = 0.f;
    if (blockIdx.x * 512 + 512 <= rows) {    // block-uniform fast path
        const float* p = Y + (long long)r0 * 128 + col;
        for (int i = 0; i < 256; ++i) {
            float v = p[(long long)i * 256];
            s += v;
            q += v * v;
        }
    } else {
        int r = r0;
        for (int i = 0; i < 256; ++i, r += 2) {
            if (r < rows) {
                float v = Y[(long long)r * 128 + col];
                s += v;
                q += v * v;
            }
        }
    }
    atomicAdd(&sum[col], s);
    atomicAdd(&sumsq[col], q);
}

// 6) fold GraphNorm into per-column scale/shift:
//    var = E[x^2] - 2*ms*mu^2 + ms^2*mu^2 ;  y = a*x + c, a=gamma*rsqrt(var+eps), c=beta-a*ms*mu
__global__ void finalize_stats_kernel(const float* __restrict__ sum,
                                      const float* __restrict__ sumsq,
                                      const float* __restrict__ gamma,
                                      const float* __restrict__ beta,
                                      const float* __restrict__ ms, float rows,
                                      float* __restrict__ scale, float* __restrict__ shift) {
    int c = threadIdx.x;
    float mu  = sum[c] / rows;
    float ex2 = sumsq[c] / rows;
    float m   = ms[c];
    float var = ex2 - 2.f * m * mu * mu + m * m * mu * mu;
    float a   = gamma[c] * rsqrtf(var + 1e-5f);
    scale[c] = a;
    shift[c] = beta[c] - a * m * mu;
}

// 7) in-place normalize + ReLU, float4 vectorized (32 float4 per row)
__global__ __launch_bounds__(256) void norm_relu_kernel(float* __restrict__ Y,
                                                        const float* __restrict__ scale,
                                                        const float* __restrict__ shift,
                                                        long long n4) {
    long long i = (long long)blockIdx.x * 256 + threadIdx.x;
    if (i >= n4) return;
    int c4 = (int)(i & 31);
    float4 v = ((float4*)Y)[i];
    float4 a = ((const float4*)scale)[c4];
    float4 b = ((const float4*)shift)[c4];
    v.x = fmaxf(0.f, a.x * v.x + b.x);
    v.y = fmaxf(0.f, a.y * v.y + b.y);
    v.z = fmaxf(0.f, a.z * v.z + b.z);
    v.w = fmaxf(0.f, a.w * v.w + b.w);
    ((float4*)Y)[i] = v;
}

// 8) scatter-mean accumulation: acc[c2o[i]] += local_h[i], cnt[c2o[i]] += 1
__global__ __launch_bounds__(256) void scatter_acc_kernel(const float* __restrict__ LH,
                                                          const int* __restrict__ c2o,
                                                          float* __restrict__ acc,
                                                          float* __restrict__ cnt, int m) {
    long long t = (long long)blockIdx.x * 256 + threadIdx.x;
    int i = (int)(t >> 5);
    int lane = threadIdx.x & 31;
    if (i >= m) return;
    int dst = c2o[i];
    float4 v = ((const float4*)(LH + (long long)i * 128))[lane];
    float* a = acc + (long long)dst * 128 + lane * 4;
    atomicAdd(a + 0, v.x);
    atomicAdd(a + 1, v.y);
    atomicAdd(a + 2, v.z);
    atomicAdd(a + 3, v.w);
    if (lane == 0) atomicAdd(&cnt[dst], 1.0f);
}

// 9) local_out = ALPHA*local_h + (1-ALPHA)*base_h[c2o]   (in place over local region)
__global__ __launch_bounds__(256) void local_out_kernel(float* __restrict__ L,
                                                        const float* __restrict__ B,
                                                        const int* __restrict__ c2o, int m) {
    long long t = (long long)blockIdx.x * 256 + threadIdx.x;
    int i = (int)(t >> 5);
    int lane = threadIdx.x & 31;
    if (i >= m) return;
    int src = c2o[i];
    float4 lh = ((const float4*)(L + (long long)i * 128))[lane];
    float4 bh = ((const float4*)(B + (long long)src * 128))[lane];
    float4 o;
    o.x = ALPHA * lh.x + (1.f - ALPHA) * bh.x;
    o.y = ALPHA * lh.y + (1.f - ALPHA) * bh.y;
    o.z = ALPHA * lh.z + (1.f - ALPHA) * bh.z;
    o.w = ALPHA * lh.w + (1.f - ALPHA) * bh.w;
    ((float4*)(L + (long long)i * 128))[lane] = o;
}

// 10) base_out = ALPHA*base_h + (1-ALPHA)*acc/max(cnt,1)   (in place over base region)
__global__ __launch_bounds__(256) void base_out_kernel(float* __restrict__ B,
                                                       const float* __restrict__ acc,
                                                       const float* __restrict__ cnt, int n) {
    long long t = (long long)blockIdx.x * 256 + threadIdx.x;
    int i = (int)(t >> 5);
    int lane = threadIdx.x & 31;
    if (i >= n) return;
    float inv = (1.f - ALPHA) / fmaxf(cnt[i], 1.0f);
    float4 bh = ((const float4*)(B + (long long)i * 128))[lane];
    float4 av = ((const float4*)(acc + (long long)i * 128))[lane];
    float4 o;
    o.x = ALPHA * bh.x + inv * av.x;
    o.y = ALPHA * bh.y + inv * av.y;
    o.z = ALPHA * bh.z + inv * av.z;
    o.w = ALPHA * bh.w + inv * av.w;
    ((float4*)(B + (long long)i * 128))[lane] = o;
}

extern "C" void kernel_launch(void* const* d_in, const int* in_sizes, int n_in,
                              void* d_out, int out_size, void* d_ws, size_t ws_size,
                              hipStream_t stream) {
    const float* base_x   = (const float*)d_in[0];
    const int*   base_ei  = (const int*)d_in[1];
    const float* base_ew  = (const float*)d_in[2];
    const float* local_x  = (const float*)d_in[3];
    const int*   local_ei = (const int*)d_in[4];
    const float* local_ew = (const float*)d_in[5];
    const int*   c2o      = (const int*)d_in[6];
    const float* base_W   = (const float*)d_in[7];
    const float* base_b   = (const float*)d_in[8];
    const float* local_W  = (const float*)d_in[9];
    const float* local_b  = (const float*)d_in[10];
    const float* bg_gamma = (const float*)d_in[11];
    const float* bg_beta  = (const float*)d_in[12];
    const float* bg_ms    = (const float*)d_in[13];
    const float* lg_gamma = (const float*)d_in[14];
    const float* lg_beta  = (const float*)d_in[15];
    const float* lg_ms    = (const float*)d_in[16];

    const int n  = in_sizes[0] / 128;
    const int m  = in_sizes[3] / 128;
    const int Eb = in_sizes[2];
    const int El = in_sizes[5];

    // workspace layout (floats)
    float* ws = (float*)d_ws;
    size_t off = 0;
    float* dinv_b = ws + off; off += (size_t)n;
    float* dinv_l = ws + off; off += (size_t)m;
    float* m_b    = ws + off; off += (size_t)n * 128;
    float* m_l    = ws + off; off += (size_t)m * 128;
    float* acc    = ws + off; off += (size_t)n * 128;
    float* cnt    = ws + off; off += (size_t)n;
    float* stats  = ws + off; off += 8 * 128;
    // stats: [0]=sum_b [128]=sq_b [256]=sum_l [384]=sq_l
    //        [512]=scale_b [640]=shift_b [768]=scale_l [896]=shift_l

    float* out_b = (float*)d_out;
    float* out_l = out_b + (size_t)n * 128;

    hipMemsetAsync(dinv_b, 0, (size_t)n * sizeof(float), stream);
    hipMemsetAsync(dinv_l, 0, (size_t)m * sizeof(float), stream);
    hipMemsetAsync(m_b, 0, (size_t)n * 128 * sizeof(float), stream);
    hipMemsetAsync(m_l, 0, (size_t)m * 128 * sizeof(float), stream);
    hipMemsetAsync(acc, 0, (size_t)n * 128 * sizeof(float), stream);
    hipMemsetAsync(cnt, 0, (size_t)n * sizeof(float), stream);
    hipMemsetAsync(stats, 0, 8 * 128 * sizeof(float), stream);

    deg_kernel<<<(Eb + 255) / 256, 256, 0, stream>>>(base_ei, base_ew, dinv_b, Eb);
    deg_kernel<<<(El + 255) / 256, 256, 0, stream>>>(local_ei, local_ew, dinv_l, El);
    dinv_kernel<<<(n + 255) / 256, 256, 0, stream>>>(dinv_b, n);
    dinv_kernel<<<(m + 255) / 256, 256, 0, stream>>>(dinv_l, m);

    spmm_kernel<<<(Eb + 7) / 8, 256, 0, stream>>>(base_ei, base_ei + Eb, base_ew,
                                                  dinv_b, base_x, m_b, Eb);
    spmm_kernel<<<(El + 7) / 8, 256, 0, stream>>>(local_ei, local_ei + El, local_ew,
                                                  dinv_l, local_x, m_l, El);

    gemm_bias_wmma_kernel<<<(n + 63) / 64, 256, 0, stream>>>(m_b, base_W, base_b, out_b, n);
    gemm_bias_wmma_kernel<<<(m + 63) / 64, 256, 0, stream>>>(m_l, local_W, local_b, out_l, m);

    colstats_kernel<<<(n + 511) / 512, 256, 0, stream>>>(out_b, n, stats + 0, stats + 128);
    colstats_kernel<<<(m + 511) / 512, 256, 0, stream>>>(out_l, m, stats + 256, stats + 384);
    finalize_stats_kernel<<<1, 128, 0, stream>>>(stats + 0, stats + 128, bg_gamma, bg_beta,
                                                 bg_ms, (float)n, stats + 512, stats + 640);
    finalize_stats_kernel<<<1, 128, 0, stream>>>(stats + 256, stats + 384, lg_gamma, lg_beta,
                                                 lg_ms, (float)m, stats + 768, stats + 896);

    long long n4b = (long long)n * 32, n4l = (long long)m * 32;
    norm_relu_kernel<<<(unsigned)((n4b + 255) / 256), 256, 0, stream>>>(out_b, stats + 512,
                                                                        stats + 640, n4b);
    norm_relu_kernel<<<(unsigned)((n4l + 255) / 256), 256, 0, stream>>>(out_l, stats + 768,
                                                                        stats + 896, n4l);

    scatter_acc_kernel<<<(m + 7) / 8, 256, 0, stream>>>(out_l, c2o, acc, cnt, m);
    local_out_kernel<<<(m + 7) / 8, 256, 0, stream>>>(out_l, out_b, c2o, m);  // reads base first
    base_out_kernel<<<(n + 7) / 8, 256, 0, stream>>>(out_b, acc, cnt, n);     // then overwrite base
}
